// Quant_DenseLayer_50337016709735
// MI455X (gfx1250) — matrix operand order repaired and checked by reading, not compiled, not run on hardware
//
#include <hip/hip_runtime.h>
#include <hip/hip_bf16.h>

typedef __attribute__((ext_vector_type(16))) _Float16 v16h;
typedef __attribute__((ext_vector_type(8)))  _Float16 v8h;
typedef __attribute__((ext_vector_type(8)))  float    v8f;
typedef __attribute__((ext_vector_type(4)))  float    v4f;

static constexpr int KDIM = 4096;   // N_IN
static constexpr int NDIM = 4096;   // N_UNITS
static constexpr int MDIM = 4096;   // batch
static constexpr size_t WTOT = (size_t)KDIM * NDIM; // 16777216

// ---------------------------------------------------------------------------
// Kernel 1: per-block partial sums of |W|  (deterministic fixed-order tree)
// grid = 4096 blocks x 256 threads; each block covers 4096 floats (1024 float4)
// ---------------------------------------------------------------------------
__global__ __launch_bounds__(256) void abs_partial(const float* __restrict__ W,
                                                   float* __restrict__ partial) {
    __shared__ float red[256];
    const int t = threadIdx.x;
    const size_t base4 = (size_t)blockIdx.x * 1024;   // in float4 units
    const v4f* W4 = (const v4f*)W;
    float s = 0.0f;
    #pragma unroll
    for (int j = 0; j < 4; ++j) {
        v4f v = W4[base4 + (size_t)j * 256 + t];
        s += fabsf(v.x) + fabsf(v.y) + fabsf(v.z) + fabsf(v.w);
    }
    red[t] = s;
    __syncthreads();
    for (int off = 128; off > 0; off >>= 1) {
        if (t < off) red[t] += red[t + off];
        __syncthreads();
    }
    if (t == 0) partial[blockIdx.x] = red[0];
}

// ---------------------------------------------------------------------------
// Kernel 2: fold 4096 partials -> threshold = 0.7 * mean(|W|)
// single block, fixed-order tree => deterministic
// ---------------------------------------------------------------------------
__global__ __launch_bounds__(256) void reduce_thresh(const float* __restrict__ partial,
                                                     float* __restrict__ thr) {
    __shared__ float red[256];
    const int t = threadIdx.x;
    float s = 0.0f;
    #pragma unroll
    for (int j = 0; j < 16; ++j) s += partial[(size_t)j * 256 + t];
    red[t] = s;
    __syncthreads();
    for (int off = 128; off > 0; off >>= 1) {
        if (t < off) red[t] += red[t + off];
        __syncthreads();
    }
    if (t == 0) thr[0] = 0.7f * (red[0] / (float)WTOT);
}

// ---------------------------------------------------------------------------
// Kernel 3: ternary-quantize W and store TRANSPOSED as f16: WqT[n][k]
// Matches reference: sign(sign(W+t)+sign(W-t))
// block (32,8), 32x32 tile via LDS, grid (NDIM/32, KDIM/32)
// ---------------------------------------------------------------------------
__global__ __launch_bounds__(256) void quantize_transpose(const float* __restrict__ W,
                                                          const float* __restrict__ thr,
                                                          _Float16* __restrict__ WqT) {
    __shared__ _Float16 tile[32][33];
    const float t = thr[0];
    const int n0 = blockIdx.x * 32;
    const int k0 = blockIdx.y * 32;
    const int tx = threadIdx.x;     // 0..31
    const int ty = threadIdx.y;     // 0..7
    #pragma unroll
    for (int i = 0; i < 32; i += 8) {
        const int k = k0 + ty + i;
        const int n = n0 + tx;
        const float w = W[(size_t)k * NDIM + n];
        const float a = w + t;
        const float c = w - t;
        const int s = ((a > 0.0f) - (a < 0.0f)) + ((c > 0.0f) - (c < 0.0f));
        const float q = (float)((s > 0) - (s < 0));
        tile[ty + i][tx] = (_Float16)q;
    }
    __syncthreads();
    #pragma unroll
    for (int i = 0; i < 32; i += 8) {
        const int n = n0 + ty + i;
        const int k = k0 + tx;
        WqT[(size_t)n * KDIM + k] = tile[tx][ty + i];
    }
}

// ---------------------------------------------------------------------------
// Kernel 4: x fp32 -> f16, vectorized (float4 in, 4 halves out)
// ---------------------------------------------------------------------------
__global__ __launch_bounds__(256) void x_to_f16(const float* __restrict__ x,
                                                _Float16* __restrict__ xh) {
    const size_t i4 = (size_t)blockIdx.x * 256 + threadIdx.x;
    v4f v = ((const v4f*)x)[i4];
    typedef __attribute__((ext_vector_type(4))) _Float16 v4h;
    v4h o;
    o.x = (_Float16)v.x; o.y = (_Float16)v.y; o.z = (_Float16)v.z; o.w = (_Float16)v.w;
    ((v4h*)xh)[i4] = o;
}

// ---------------------------------------------------------------------------
// Fragment load helpers (16-bit A/B fragments, contiguous k order: lane half h holds K 16h..16h+15)
// ---------------------------------------------------------------------------
__device__ __forceinline__ v16h load_afrag(const _Float16* aBase, int mi, int k) {
    // lane holds row (lm); VGPR0..7 = K lh*16..+15 (contiguous)
    const _Float16* p = aBase + (size_t)mi * 16 * KDIM + k;
    union { v16h v; v8h h[2]; } u;
    u.h[0] = *(const v8h*)(p);
    u.h[1] = *(const v8h*)(p + 8);
    return u.v;
}

__device__ __forceinline__ v16h load_bfrag(const _Float16* bBase, int ni, int k) {
    // lanes 0-15: K 0-15, lanes 16-31: K 16-31 (lh*16 folded into bBase); contiguous
    return *(const v16h*)(bBase + (size_t)ni * 16 * KDIM + k);
}

// ---------------------------------------------------------------------------
// Kernel 5: WMMA GEMM  out[M][N] = Xh[M][K] * WqT[N][K]^T + b
// 8 waves/block, workgroup tile 128x128, wave tile 64x32 (4x2 WMMA tiles).
// Software-pipelined: fragments for k+32 are issued before the WMMAs for k,
// so s_wait_loadcnt never has to drain to 0 inside the steady-state loop.
// ---------------------------------------------------------------------------
__global__ __launch_bounds__(256) void ternary_gemm(const _Float16* __restrict__ Xh,
                                                    const _Float16* __restrict__ WqT,
                                                    const float* __restrict__ bias,
                                                    float* __restrict__ out) {
    const int lane  = threadIdx.x & 31;
    const int wave  = threadIdx.x >> 5;
    const int waveM = wave & 1;         // 0..1  -> 64-row band
    const int waveN = wave >> 1;        // 0..3  -> 32-col band
    const int rowBase = blockIdx.x * 128 + waveM * 64;
    const int colBase = blockIdx.y * 128 + waveN * 32;
    const int lm = lane & 15;           // A: row-in-tile / B: col-in-tile / D: col
    const int lh = lane >> 4;           // K half-group selector

    v8f acc[4][2] = {};

    const _Float16* aBase = Xh + (size_t)(rowBase + lm) * KDIM + (size_t)lh * 16;
    const _Float16* bBase = WqT + (size_t)(colBase + lm) * KDIM + (size_t)lh * 16;

    v16h a0[4], a1[4];
    v16h b0[2], b1[2];

    // prologue: fragments for k = 0
    #pragma unroll
    for (int mi = 0; mi < 4; ++mi) a0[mi] = load_afrag(aBase, mi, 0);
    #pragma unroll
    for (int ni = 0; ni < 2; ++ni) b0[ni] = load_bfrag(bBase, ni, 0);

    #define WMMA_BLOCK(A, B)                                                    \
        _Pragma("unroll")                                                       \
        for (int mi = 0; mi < 4; ++mi) {                                        \
            _Pragma("unroll")                                                   \
            for (int ni = 0; ni < 2; ++ni) {                                    \
                acc[mi][ni] = __builtin_amdgcn_wmma_f32_16x16x32_f16(           \
                    false, (A)[mi], false, (B)[ni],                             \
                    (short)0, acc[mi][ni], false, false);                       \
            }                                                                   \
        }

    // steady state: K unrolled by 64, two rotating fragment buffers
    for (int k = 0; k < KDIM - 64; k += 64) {
        #pragma unroll
        for (int mi = 0; mi < 4; ++mi) a1[mi] = load_afrag(aBase, mi, k + 32);
        #pragma unroll
        for (int ni = 0; ni < 2; ++ni) b1[ni] = load_bfrag(bBase, ni, k + 32);
        WMMA_BLOCK(a0, b0)

        #pragma unroll
        for (int mi = 0; mi < 4; ++mi) a0[mi] = load_afrag(aBase, mi, k + 64);
        #pragma unroll
        for (int ni = 0; ni < 2; ++ni) b0[ni] = load_bfrag(bBase, ni, k + 64);
        WMMA_BLOCK(a1, b1)
    }

    // epilogue: last 64 K (k = KDIM-64); all loads in bounds
    {
        #pragma unroll
        for (int mi = 0; mi < 4; ++mi) a1[mi] = load_afrag(aBase, mi, KDIM - 32);
        #pragma unroll
        for (int ni = 0; ni < 2; ++ni) b1[ni] = load_bfrag(bBase, ni, KDIM - 32);
        WMMA_BLOCK(a0, b0)
        WMMA_BLOCK(a1, b1)
    }
    #undef WMMA_BLOCK

    // D layout: VGPR v, lanes 0-15 -> M = v, lanes 16-31 -> M = 8+v; N = lane&15
    #pragma unroll
    for (int mi = 0; mi < 4; ++mi) {
        #pragma unroll
        for (int ni = 0; ni < 2; ++ni) {
            const int col = colBase + ni * 16 + lm;
            const float bv = bias[col];
            #pragma unroll
            for (int v = 0; v < 8; ++v) {
                const int row = rowBase + mi * 16 + lh * 8 + v;
                out[(size_t)row * NDIM + col] = acc[mi][ni][v] + bv;
            }
        }
    }
}

// ---------------------------------------------------------------------------
// Host-side launcher
// ---------------------------------------------------------------------------
extern "C" void kernel_launch(void* const* d_in, const int* in_sizes, int n_in,
                              void* d_out, int out_size, void* d_ws, size_t ws_size,
                              hipStream_t stream) {
    const float* x = (const float*)d_in[0];   // [4096,4096] fp32
    const float* W = (const float*)d_in[1];   // [4096,4096] fp32
    const float* b = (const float*)d_in[2];   // [4096]      fp32
    float* out = (float*)d_out;               // [4096,4096] fp32

    // Workspace layout: Xh (32MB) | WqT (32MB) | partials (16KB) | thr (4B)
    char* ws = (char*)d_ws;
    _Float16* Xh     = (_Float16*)ws;
    _Float16* WqT    = (_Float16*)(ws + (size_t)MDIM * KDIM * 2);
    float*    part   = (float*)(ws + (size_t)MDIM * KDIM * 2 + (size_t)KDIM * NDIM * 2);
    float*    thr    = part + 4096;

    // 1) |W| partial sums: 4096 blocks cover 16.7M floats
    abs_partial<<<4096, 256, 0, stream>>>(W, part);
    // 2) threshold = 0.7 * mean(|W|)
    reduce_thresh<<<1, 256, 0, stream>>>(part, thr);
    // 3) ternary quantize + transpose -> WqT[n][k] in f16
    {
        dim3 grid(NDIM / 32, KDIM / 32);
        dim3 block(32, 8);
        quantize_transpose<<<grid, block, 0, stream>>>(W, thr, WqT);
    }
    // 4) x -> f16
    {
        const int nBlocks = (int)((size_t)MDIM * KDIM / 4 / 256); // 16384
        x_to_f16<<<nBlocks, 256, 0, stream>>>(x, Xh);
    }
    // 5) WMMA GEMM + bias
    {
        dim3 grid(MDIM / 128, NDIM / 128);  // 32 x 32
        ternary_gemm<<<grid, 256, 0, stream>>>(Xh, WqT, b, out);
    }
}
